// ProtGCNBlock_47502338293794
// MI455X (gfx1250) — compile-verified, matrix operand order checked by reading, not run on hardware
//
#include <hip/hip_runtime.h>
#include <hip/hip_bf16.h>
#include <math.h>

// ---------------------------------------------------------------------------
// ProtGCN forward for MI455X (gfx1250, wave32).
//
// Pipeline is HBM-bandwidth bound (esm_emb = 512MB, e-tensors ~600MB/layer,
// scatter ~1.8GB/layer vs ~180 GFLOP of GEMM). GEMMs use
// v_wmma_f32_16x16x32_bf16 (f32 accumulate), converting fp32->bf16 only while
// staging tiles into LDS. The [seq|node_s|esm_emb] concat is fused into a
// dedicated GEMM kernel so esm_emb is read exactly once. e_l (the only big
// zero-reuse intermediate) is stored bf16 to halve its HBM traffic.
// q/k/v/x (~51MB each) fit the 192MB L2, so per-edge gathers are L2-resident.
//
// Input pointer layout assumption (setup_inputs() dict insertion order,
// params flattened recursively in insertion order):
//   0 seq[50000,33] 1 edge_index[2,600000](i32) 2 node_s[50000,6]
//   3 esm_emb[50000,2560] 4 edge_s[600000,39] 5 batch[50000](i32)
//   6 proj_node.W[2599,128] 7 proj_node.b[128]
//   8 proj_edge.W[39,128]   9 proj_edge.b[128]
//   then per layer l=0..2 (9 each, base 10+9l):
//     q.W q.b k.W k.b v.W v.b e.W skip.W skip.b
// ---------------------------------------------------------------------------

typedef __attribute__((ext_vector_type(16))) __bf16 v16bf;
typedef __attribute__((ext_vector_type(8)))  float  v8f;

#define N_NODES  50000
#define N_EDGES  600000
#define N_GRAPHS 100

// ---------------- GEMM: C[M,N] = A[M,K] * B[K,N] (+bias) -------------------
// Block tile 64(M) x 128(N), KB=32. 256 threads = 8 waves in a 2x4 grid,
// each wave owns a 32x32 sub-tile -> 2x2 v_wmma_f32_16x16x32_bf16.

#define GT_M 64
#define GT_N 128
#define GT_K 32
#define GT_KP (GT_K + 8)   // bf16 LDS row pad (80B stride) spreads banks
#define GT_CP (GT_N + 4)   // f32 epilogue row pad

union SmemT {
  struct { __bf16 A[GT_M][GT_KP]; __bf16 B[GT_N][GT_KP]; } t;  // 15.4 KB
  float C[GT_M][GT_CP];                                        // 33.8 KB
};

__device__ __forceinline__ unsigned pack_bf16(float a, float b) {
  union { __bf16 h[2]; unsigned u; } pk;
  pk.h[0] = (__bf16)a;
  pk.h[1] = (__bf16)b;
  return pk.u;
}

// Fast A staging: full 32-wide K tile, K % 4 == 0. One thread = 8 contiguous
// K elems of one row: 2x global_load_b128 -> 4x cvt_pk -> 1x ds_store_b128.
__device__ __forceinline__ void stage_a_fast(SmemT& sm, const float* __restrict__ A,
                                             int k0, int bm, int M, int K, int tid) {
  const int r  = tid >> 2;          // 0..63
  const int kq = (tid & 3) << 3;    // 0,8,16,24
  const int gr = bm + r;
  uint4 u = {0u, 0u, 0u, 0u};
  if (gr < M) {
    const float* p = A + (size_t)gr * K + k0 + kq;
    __builtin_prefetch(p + GT_K, 0, 0);      // global_prefetch_b8 next K tile
    float4 r0 = *(const float4*)p;
    float4 r1 = *(const float4*)(p + 4);
    u.x = pack_bf16(r0.x, r0.y);
    u.y = pack_bf16(r0.z, r0.w);
    u.z = pack_bf16(r1.x, r1.y);
    u.w = pack_bf16(r1.z, r1.w);
  }
  *(uint4*)&sm.t.A[r][kq] = u;      // 80B row stride keeps 16B alignment
}

// Fast B staging (transpose into lB[n][k]): full tile, N tile fully in range.
// One thread handles 2 K-rows x 4 N-cols: 2x b128 loads -> 4x ds_store_b32.
__device__ __forceinline__ void stage_b_fast(SmemT& sm, const float* __restrict__ B,
                                             int k0, int bn, int N, int tid) {
#pragma unroll
  for (int it = 0; it < 2; ++it) {
    const int g  = tid + it * 256;
    const int kk = (g >> 5) << 1;   // 0,2,...,30
    const int nn = (g & 31) << 2;   // 0,4,...,124
    const float* p0 = B + (size_t)(k0 + kk) * N + bn + nn;
    const float* p1 = p0 + N;
    __builtin_prefetch(p0 + (size_t)GT_K * N, 0, 0);
    float4 r0 = *(const float4*)p0;
    float4 r1 = *(const float4*)p1;
    *(unsigned*)&sm.t.B[nn + 0][kk] = pack_bf16(r0.x, r1.x);
    *(unsigned*)&sm.t.B[nn + 1][kk] = pack_bf16(r0.y, r1.y);
    *(unsigned*)&sm.t.B[nn + 2][kk] = pack_bf16(r0.z, r1.z);
    *(unsigned*)&sm.t.B[nn + 3][kk] = pack_bf16(r0.w, r1.w);
  }
}

// Generic (guarded, scalar) staging for ragged K / tile tails.
__device__ __forceinline__ void stage_scalar(SmemT& sm, const float* __restrict__ A,
                                             const float* __restrict__ B, int k0,
                                             int bm, int bn, int M, int K, int N, int tid) {
  for (int i = tid; i < GT_M * GT_K; i += 256) {
    int r = i >> 5, c = i & 31;
    int gr = bm + r, gc = k0 + c;
    float v = (gr < M && gc < K) ? A[(size_t)gr * K + gc] : 0.f;
    sm.t.A[r][c] = (__bf16)v;
  }
  for (int i = tid; i < GT_N * GT_K; i += 256) {
    int r = i >> 7, c = i & 127;
    int gk = k0 + r, gn = bn + c;
    float v = (gk < K && gn < N) ? B[(size_t)gk * N + gn] : 0.f;
    sm.t.B[c][r] = (__bf16)v;
  }
}

// Fragment load per ISA 7.12.2 16-bit layout: per lane, element j -> K=8*lh+j
// (j<8) and K=8*lh+16+(j-8) (j>=8): exactly two contiguous 16B LDS chunks.
// Loaded as 2x uint4 to force ds_load_b128.
__device__ __forceinline__ v16bf load_frag(const __bf16* __restrict__ p, int lh) {
  union { uint4 q[2]; v16bf v; } u;
  u.q[0] = *(const uint4*)(p + lh * 8);
  u.q[1] = *(const uint4*)(p + lh * 8 + 16);
  return u.v;
}

__device__ __forceinline__ void wmma_step(SmemT& sm, v8f acc[2][2],
                                          int wm, int wn, int row, int lh) {
  v16bf af[2], bf[2];
#pragma unroll
  for (int i = 0; i < 2; ++i) {
    af[i] = load_frag(&sm.t.A[wm * 32 + i * 16 + row][0], lh);
    bf[i] = load_frag(&sm.t.B[wn * 32 + i * 16 + row][0], lh);
  }
#pragma unroll
  for (int i = 0; i < 2; ++i)
#pragma unroll
    for (int j = 0; j < 2; ++j)
      acc[i][j] = __builtin_amdgcn_wmma_f32_16x16x32_bf16(
          false, af[i], false, bf[j], (short)0, acc[i][j], false, false);
}

// Epilogue through LDS: coalesced float4 stores (or packed-bf16 uint2 stores
// when Cbf != nullptr). C/D layout: elem r -> row (r+8*lh), col = lane&15.
__device__ __forceinline__ void epilogue(SmemT& sm, v8f acc[2][2],
                                         const float* __restrict__ bias,
                                         float* __restrict__ C,
                                         __bf16* __restrict__ Cbf,
                                         int bm, int bn, int M, int N, int tid,
                                         int wm, int wn, int lane, int lh) {
  __syncthreads();   // LDS union: tiles no longer needed
#pragma unroll
  for (int i = 0; i < 2; ++i)
#pragma unroll
    for (int j = 0; j < 2; ++j)
#pragma unroll
      for (int r = 0; r < 8; ++r)
        sm.C[wm * 32 + i * 16 + lh * 8 + r][wn * 32 + j * 16 + (lane & 15)] =
            acc[i][j][r];
  __syncthreads();
  for (int idx = tid; idx < GT_M * (GT_N >> 2); idx += 256) {
    int r  = idx >> 5;
    int c4 = (idx & 31) << 2;
    int gr = bm + r, gc = bn + c4;
    if (gr >= M || gc + 4 > N) continue;
    float4 v;
    v.x = sm.C[r][c4 + 0];
    v.y = sm.C[r][c4 + 1];
    v.z = sm.C[r][c4 + 2];
    v.w = sm.C[r][c4 + 3];
    if (bias) {
      v.x += bias[gc + 0]; v.y += bias[gc + 1];
      v.z += bias[gc + 2]; v.w += bias[gc + 3];
    }
    if (Cbf) {
      uint2 pk;
      pk.x = pack_bf16(v.x, v.y);
      pk.y = pack_bf16(v.z, v.w);
      *(uint2*)&Cbf[(size_t)gr * N + gc] = pk;
    } else {
      *(float4*)&C[(size_t)gr * N + gc] = v;
    }
  }
}

// Plain GEMM (all layer GEMMs + edge projection).
__global__ __launch_bounds__(256)
void gemm_bf16_wmma(const float* __restrict__ A, const float* __restrict__ B,
                    const float* __restrict__ bias, float* __restrict__ C,
                    __bf16* __restrict__ Cbf, int M, int K, int N) {
  __shared__ SmemT sm;
  const int tid  = threadIdx.x;
  const int wave = tid >> 5, lane = tid & 31;
  const int wm   = wave >> 2, wn = wave & 3;
  const int row  = lane & 15, lh = lane >> 4;
  const int bm   = blockIdx.y * GT_M, bn = blockIdx.x * GT_N;
  const bool fast = ((K & 31) == 0) && ((N & 127) == 0);   // uniform

  v8f acc[2][2] = {};
  for (int k0 = 0; k0 < K; k0 += GT_K) {
    if (fast) {
      stage_a_fast(sm, A, k0, bm, M, K, tid);
      stage_b_fast(sm, B, k0, bn, N, tid);
    } else {
      stage_scalar(sm, A, B, k0, bm, bn, M, K, N, tid);
    }
    __syncthreads();
    wmma_step(sm, acc, wm, wn, row, lh);
    __syncthreads();
  }
  epilogue(sm, acc, bias, C, Cbf, bm, bn, M, N, tid, wm, wn, lane, lh);
}

// Concat GEMM for proj_node: A = [seq(33) | node_s(6) | esm(2560)], K = 2599.
// For k0 >= 64 (79 of 82 tiles) every column is in esm -> branch-free
// streaming stage; ragged first/last tiles use guarded scalar staging.
__global__ __launch_bounds__(256)
void gemm_bf16_wmma_concat(const float* __restrict__ seq,
                           const float* __restrict__ ns,
                           const float* __restrict__ esm,
                           const float* __restrict__ B,
                           const float* __restrict__ bias,
                           float* __restrict__ C, int M, int N) {
  const int K = 2599;
  __shared__ SmemT sm;
  const int tid  = threadIdx.x;
  const int wave = tid >> 5, lane = tid & 31;
  const int wm   = wave >> 2, wn = wave & 3;
  const int row  = lane & 15, lh = lane >> 4;
  const int bm   = blockIdx.y * GT_M, bn = blockIdx.x * GT_N;

  v8f acc[2][2] = {};
  for (int k0 = 0; k0 < K; k0 += GT_K) {
    const bool full = (k0 + GT_K <= K);                    // uniform
    if (full && k0 >= 64) {
      // pure-esm tile: coalesced scalar loads (esm rows unaligned for b128),
      // packed to bf16, single ds_store_b128 per thread.
      const int r  = tid >> 2;
      const int kq = (tid & 3) << 3;
      const int gr = bm + r;
      uint4 u = {0u, 0u, 0u, 0u};
      if (gr < M) {
        const float* p = esm + (size_t)gr * 2560 + (k0 - 39) + kq;
        __builtin_prefetch(p + GT_K, 0, 0);
        float a[8];
#pragma unroll
        for (int j = 0; j < 8; ++j) a[j] = p[j];
        u.x = pack_bf16(a[0], a[1]);
        u.y = pack_bf16(a[2], a[3]);
        u.z = pack_bf16(a[4], a[5]);
        u.w = pack_bf16(a[6], a[7]);
      }
      *(uint4*)&sm.t.A[r][kq] = u;
    } else {
      for (int i = tid; i < GT_M * GT_K; i += 256) {
        int r = i >> 5, c = i & 31;
        int gr = bm + r, gc = k0 + c;
        float v = 0.f;
        if (gr < M && gc < K) {
          if (gc < 33)      v = seq[(size_t)gr * 33 + gc];
          else if (gc < 39) v = ns[(size_t)gr * 6 + (gc - 33)];
          else              v = esm[(size_t)gr * 2560 + (gc - 39)];
        }
        sm.t.A[r][c] = (__bf16)v;
      }
    }
    if (full) {
      stage_b_fast(sm, B, k0, bn, N, tid);
    } else {
      for (int i = tid; i < GT_N * GT_K; i += 256) {
        int r = i >> 7, c = i & 127;
        int gk = k0 + r, gn = bn + c;
        float v = (gk < K && gn < N) ? B[(size_t)gk * N + gn] : 0.f;
        sm.t.B[c][r] = (__bf16)v;
      }
    }
    __syncthreads();
    wmma_step(sm, acc, wm, wn, row, lh);
    __syncthreads();
  }
  epilogue(sm, acc, bias, C, nullptr, bm, bn, M, N, tid, wm, wn, lane, lh);
}

// ---------------------- attention / softmax passes -------------------------

__device__ __forceinline__ void atomic_max_f32(float* addr, float val) {
  unsigned int* ai = (unsigned int*)addr;
  unsigned int old = *ai;
  while (__uint_as_float(old) < val) {
    unsigned int assumed = old;
    old = atomicCAS(ai, assumed, __float_as_uint(val));
    if (old == assumed) break;
  }
}

__global__ void init_ms_kernel(float* __restrict__ m, float* __restrict__ s, int n) {
  int i = blockIdx.x * blockDim.x + threadIdx.x;
  if (i < n) { m[i] = -INFINITY; s[i] = 0.f; }
}

// pass 1: alpha[e] = dot(q[dst], k[src]+e_l[e]) * scale; segment max (wave/edge)
__global__ __launch_bounds__(256)
void edge_alpha_kernel(const float* __restrict__ q, const float* __restrict__ k,
                       const __bf16* __restrict__ el, const int* __restrict__ src,
                       const int* __restrict__ dst, float* __restrict__ alpha,
                       float* __restrict__ mmax, int E, int d, float scale) {
  int e    = (int)((blockIdx.x * (unsigned)blockDim.x + threadIdx.x) >> 5);
  int lane = threadIdx.x & 31;
  if (e >= E) return;
  int sN = src[e], dN = dst[e];
  const float*  qp = q  + (size_t)dN * d;
  const float*  kp = k  + (size_t)sN * d;
  const __bf16* ep = el + (size_t)e  * d;
  float acc = 0.f;
  for (int f = lane; f < d; f += 32) acc += qp[f] * (kp[f] + (float)ep[f]);
#pragma unroll
  for (int off = 16; off > 0; off >>= 1) acc += __shfl_xor(acc, off, 32);
  acc *= scale;
  if (lane == 0) { alpha[e] = acc; atomic_max_f32(&mmax[dN], acc); }
}

// pass 2: a = exp(alpha - m[dst]); segment sum
__global__ void edge_exp_kernel(const float* __restrict__ alpha,
                                const int* __restrict__ dst,
                                const float* __restrict__ mmax,
                                float* __restrict__ av, float* __restrict__ ssum,
                                int E) {
  int e = blockIdx.x * blockDim.x + threadIdx.x;
  if (e >= E) return;
  int dN = dst[e];
  float a = __expf(alpha[e] - mmax[dN]);
  av[e] = a;
  atomicAdd(&ssum[dN], a);
}

// pass 3: out[dst] += (v[src]+e_l[e]) * a/(s[dst]+eps)  (wave per edge)
__global__ __launch_bounds__(256)
void edge_scatter_kernel(const float* __restrict__ v, const __bf16* __restrict__ el,
                         const int* __restrict__ src, const int* __restrict__ dst,
                         const float* __restrict__ av, const float* __restrict__ ssum,
                         float* __restrict__ out, int E, int d) {
  int e    = (int)((blockIdx.x * (unsigned)blockDim.x + threadIdx.x) >> 5);
  int lane = threadIdx.x & 31;
  if (e >= E) return;
  int sN = src[e], dN = dst[e];
  float w = av[e] / (ssum[dN] + 1e-16f);
  const float*  vp = v  + (size_t)sN * d;
  const __bf16* ep = el + (size_t)e  * d;
  float*        op = out + (size_t)dN * d;
  for (int f = lane; f < d; f += 32) atomicAdd(&op[f], (vp[f] + (float)ep[f]) * w);
}

__global__ void leaky_relu_kernel(float* __restrict__ x, long n) {
  long i = blockIdx.x * (long)blockDim.x + threadIdx.x;
  if (i >= n) return;
  float v = x[i];
  x[i] = v > 0.f ? v : 0.01f * v;
}

// ------------------------------ mean pool ----------------------------------

__global__ void zero_f32_kernel(float* __restrict__ p, long n) {
  long i = blockIdx.x * (long)blockDim.x + threadIdx.x;
  if (i < n) p[i] = 0.f;
}

__global__ void pool_sum_kernel(const float* __restrict__ x,
                                const int* __restrict__ batch,
                                float* __restrict__ sums, int n, int d) {
  long i = blockIdx.x * (long)blockDim.x + threadIdx.x;
  if (i >= (long)n * d) return;
  int node = (int)(i / d), f = (int)(i % d);
  atomicAdd(&sums[(size_t)batch[node] * d + f], x[i]);
}

__global__ void pool_cnt_kernel(const int* __restrict__ batch,
                                float* __restrict__ cnt, int n) {
  int i = blockIdx.x * blockDim.x + threadIdx.x;
  if (i < n) atomicAdd(&cnt[batch[i]], 1.f);
}

__global__ void pool_div_kernel(const float* __restrict__ sums,
                                const float* __restrict__ cnt,
                                float* __restrict__ out, int g, int d) {
  int i = blockIdx.x * blockDim.x + threadIdx.x;
  if (i >= g * d) return;
  out[i] = sums[i] / fmaxf(cnt[i / d], 1.f);
}

// ------------------------------ launcher -----------------------------------

extern "C" void kernel_launch(void* const* d_in, const int* in_sizes, int n_in,
                              void* d_out, int out_size, void* d_ws, size_t ws_size,
                              hipStream_t stream) {
  const int N = N_NODES, E = N_EDGES, G = N_GRAPHS;

  const float* seq    = (const float*)d_in[0];
  const int*   ei     = (const int*)  d_in[1];
  const float* node_s = (const float*)d_in[2];
  const float* esm    = (const float*)d_in[3];
  const float* edge_s = (const float*)d_in[4];
  const int*   batch  = (const int*)  d_in[5];
  const int*   srcI   = ei;
  const int*   dstI   = ei + E;

  const float* pnW = (const float*)d_in[6];
  const float* pnB = (const float*)d_in[7];
  const float* peW = (const float*)d_in[8];
  const float* peB = (const float*)d_in[9];

  struct Layer { const float *qW,*qb,*kW,*kb,*vW,*vb,*eW,*sW,*sb; };
  Layer L[3];
  {
    int p = 10;
    for (int i = 0; i < 3; ++i) {
      L[i].qW = (const float*)d_in[p++]; L[i].qb = (const float*)d_in[p++];
      L[i].kW = (const float*)d_in[p++]; L[i].kb = (const float*)d_in[p++];
      L[i].vW = (const float*)d_in[p++]; L[i].vb = (const float*)d_in[p++];
      L[i].eW = (const float*)d_in[p++];                       // no bias
      L[i].sW = (const float*)d_in[p++]; L[i].sb = (const float*)d_in[p++];
    }
  }

  // ---- carve workspace (~0.9 GB) ----
  char* w = (char*)d_ws;
  auto carve = [&](size_t bytes) {
    char* r = w;
    w += (bytes + 255) & ~(size_t)255;
    return r;
  };
  float*  xa    = (float*)carve((size_t)N * 256 * 4);
  float*  xb    = (float*)carve((size_t)N * 256 * 4);
  float*  qbuf  = (float*)carve((size_t)N * 256 * 4);
  float*  kbuf  = (float*)carve((size_t)N * 256 * 4);
  float*  vbuf  = (float*)carve((size_t)N * 256 * 4);
  float*  ea    = (float*)carve((size_t)E * 128 * 4);
  __bf16* el    = (__bf16*)carve((size_t)E * 256 * 2);   // bf16: zero-reuse stream
  float*  alpha = (float*)carve((size_t)E * 4);
  float*  av    = (float*)carve((size_t)E * 4);
  float*  mx    = (float*)carve((size_t)N * 4);
  float*  ss    = (float*)carve((size_t)N * 4);
  float*  sums  = (float*)carve((size_t)G * 256 * 4);
  float*  cnt   = (float*)carve((size_t)G * 4);

  auto gemm = [&](const float* A, const float* B, const float* bias, float* C,
                  __bf16* Cbf, int M, int K, int Nn) {
    dim3 grid((Nn + GT_N - 1) / GT_N, (M + GT_M - 1) / GT_M);
    gemm_bf16_wmma<<<grid, 256, 0, stream>>>(A, B, bias, C, Cbf, M, K, Nn);
  };

  // input projections
  {
    dim3 grid(1, (N + GT_M - 1) / GT_M);
    gemm_bf16_wmma_concat<<<grid, 256, 0, stream>>>(seq, node_s, esm, pnW, pnB,
                                                    xa, N, 128);
  }
  gemm(edge_s, peW, peB, ea, nullptr, E, 39, 128);

  float* xcur = xa;
  float* xnxt = xb;
  const int din_[3]  = {128, 128, 256};
  const int dout_[3] = {128, 256, 256};

  for (int l = 0; l < 3; ++l) {
    const int din = din_[l], dout = dout_[l];
    const float scale = 1.f / sqrtf((float)dout);

    gemm(xcur, L[l].qW, L[l].qb, qbuf, nullptr, N, din, dout);
    gemm(xcur, L[l].kW, L[l].kb, kbuf, nullptr, N, din, dout);
    gemm(xcur, L[l].vW, L[l].vb, vbuf, nullptr, N, din, dout);
    gemm(ea,   L[l].eW, nullptr, nullptr, el,   E, 128, dout);   // bf16 out
    gemm(xcur, L[l].sW, L[l].sb, xnxt, nullptr, N, din, dout);   // skip

    init_ms_kernel<<<(N + 255) / 256, 256, 0, stream>>>(mx, ss, N);
    edge_alpha_kernel<<<(E + 7) / 8, 256, 0, stream>>>(
        qbuf, kbuf, el, srcI, dstI, alpha, mx, E, dout, scale);
    edge_exp_kernel<<<(E + 255) / 256, 256, 0, stream>>>(
        alpha, dstI, mx, av, ss, E);
    edge_scatter_kernel<<<(E + 7) / 8, 256, 0, stream>>>(
        vbuf, el, srcI, dstI, av, ss, xnxt, E, dout);

    long nelem = (long)N * dout;
    leaky_relu_kernel<<<(int)((nelem + 255) / 256), 256, 0, stream>>>(xnxt, nelem);

    float* t = xcur; xcur = xnxt; xnxt = t;
  }

  // global mean pool over batch -> [100, 256]
  zero_f32_kernel<<<((long)G * 256 + 255) / 256, 256, 0, stream>>>(sums, (long)G * 256);
  zero_f32_kernel<<<1, 256, 0, stream>>>(cnt, G);
  pool_sum_kernel<<<(int)(((long)N * 256 + 255) / 256), 256, 0, stream>>>(
      xcur, batch, sums, N, 256);
  pool_cnt_kernel<<<(N + 255) / 256, 256, 0, stream>>>(batch, cnt, N);
  pool_div_kernel<<<(G * 256 + 255) / 256, 256, 0, stream>>>(
      sums, cnt, (float*)d_out, G, 256);
}
